// SparseLinearAttention_4440996184228
// MI455X (gfx1250) — compile-verified, matrix operand order checked by reading, not run on hardware
//
#include <hip/hip_runtime.h>
#include <hip/hip_bf16.h>

typedef __attribute__((ext_vector_type(16))) _Float16 v16h;
typedef __attribute__((ext_vector_type(8)))  float    v8f;

namespace sla {

constexpr int B = 2, L = 4096, H = 16, D = 128;
constexpr int BLKQ = 64, BLKK = 64;
constexpr int M  = L / BLKQ;   // 64 query blocks
constexpr int KB = L / BLKK;   // 64 kv blocks
constexpr int TOPK = 6;        // int(0.1 * 64)
constexpr float SCALE = 0.08838834764831845f; // D^-0.5
constexpr float EPSF  = 1e-6f;

__device__ __forceinline__ v8f wmma16(v16h a, v16h b, v8f c) {
  // v_wmma_f32_16x16x32_f16: (neg_a, A, neg_b, B, c_mod, C, reuse_a, reuse_b)
  return __builtin_amdgcn_wmma_f32_16x16x32_f16(false, a, false, b, (short)0, c,
                                                false, false);
}

__device__ __forceinline__ v8f vzero() {
  v8f z = {0.f, 0.f, 0.f, 0.f, 0.f, 0.f, 0.f, 0.f};
  return z;
}

// A fragment 16x32 f16, source row-major: elem(m,k) = s[(m0+m)*ld + k0+k]
__device__ __forceinline__ v16h load_A(const _Float16* s, int ld, int m0, int k0) {
  const int lane = threadIdx.x & 31;
  const int m = lane & 15, hh = lane >> 4;
  v16h a;
#pragma unroll
  for (int vv = 0; vv < 8; ++vv) {
    const int kk = k0 + ((vv >> 2) << 4) + (hh << 3) + ((vv & 3) << 1);
    const _Float16* p = s + (m0 + m) * ld + kk;
    a[2 * vv] = p[0];
    a[2 * vv + 1] = p[1];
  }
  return a;
}

// A fragment from transposed source: elem(m,k) = s[(k0+k)*ld + m0+m]
__device__ __forceinline__ v16h load_At(const _Float16* s, int ld, int m0, int k0) {
  const int lane = threadIdx.x & 31;
  const int m = lane & 15, hh = lane >> 4;
  v16h a;
#pragma unroll
  for (int vv = 0; vv < 8; ++vv) {
    const int kk = k0 + ((vv >> 2) << 4) + (hh << 3) + ((vv & 3) << 1);
    a[2 * vv]     = s[(size_t)kk * ld + m0 + m];
    a[2 * vv + 1] = s[(size_t)(kk + 1) * ld + m0 + m];
  }
  return a;
}

// B fragment 32x16 f16, source row-major [K][N]: elem(k,n) = s[(k0+k)*ld + n0+n]
__device__ __forceinline__ v16h load_B(const _Float16* s, int ld, int k0, int n0) {
  const int lane = threadIdx.x & 31;
  const int n = lane & 15, hh = lane >> 4;
  v16h b;
#pragma unroll
  for (int vv = 0; vv < 8; ++vv) {
    const int kk = k0 + ((vv >> 2) << 4) + (hh << 3) + ((vv & 3) << 1);
    b[2 * vv]     = s[(size_t)kk * ld + n0 + n];
    b[2 * vv + 1] = s[(size_t)(kk + 1) * ld + n0 + n];
  }
  return b;
}

// B fragment where B[k][n] = s[(n0+n)*ld + k0+k]  (e.g. K^T from row-major K)
__device__ __forceinline__ v16h load_Bt(const _Float16* s, int ld, int k0, int n0) {
  const int lane = threadIdx.x & 31;
  const int n = lane & 15, hh = lane >> 4;
  v16h b;
#pragma unroll
  for (int vv = 0; vv < 8; ++vv) {
    const int kk = k0 + ((vv >> 2) << 4) + (hh << 3) + ((vv & 3) << 1);
    const _Float16* p = s + (size_t)(n0 + n) * ld + kk;
    b[2 * vv] = p[0];
    b[2 * vv + 1] = p[1];
  }
  return b;
}

// ---------------------------------------------------------------- pooling
__global__ __launch_bounds__(128) void pool_kernel(
    const float* __restrict__ q, const float* __restrict__ k,
    float* __restrict__ pq, float* __restrict__ pk) {
  const int gid = blockIdx.x;
  const int mb = gid % M, bh = gid / M;
  const int b = bh / H, h = bh % H;
  const int d = threadIdx.x;
  float sq = 0.f, sk = 0.f;
  const size_t base = (((size_t)b * L + (size_t)mb * BLKQ) * H + h) * (size_t)D + d;
  for (int r = 0; r < BLKQ; ++r) {
    sq += q[base + (size_t)r * H * D];
    sk += k[base + (size_t)r * H * D];
  }
  pq[((size_t)bh * M + mb) * D + d] = sq * (1.f / BLKQ);
  pk[((size_t)bh * KB + mb) * D + d] = sk * (1.f / BLKK);
}

// ------------------------------------------------------- routing + top-k
__global__ __launch_bounds__(64) void route_kernel(
    const float* __restrict__ pq, const float* __restrict__ pk,
    int* __restrict__ lut) {
  __shared__ float pkc[KB * D];     // 32 KB
  __shared__ float score[M * KB];   // 16 KB
  const int bh = blockIdx.x;
  const int t = threadIdx.x;

  for (int d = 0; d < D; ++d)
    pkc[t * D + d] = pk[((size_t)bh * KB + t) * D + d];
  __syncthreads();

  // subtract per-(b,h,d) mean (mean over blocks == mean over L)
  for (int dd = t; dd < D; dd += 64) {
    float s = 0.f;
    for (int r = 0; r < KB; ++r) s += pkc[r * D + dd];
    s *= (1.f / KB);
    for (int r = 0; r < KB; ++r) pkc[r * D + dd] -= s;
  }
  __syncthreads();

  for (int mm = 0; mm < M; ++mm) {
    const float* pqr = pq + ((size_t)bh * M + mm) * D;
    float s = 0.f;
    for (int d = 0; d < D; ++d) s += pqr[d] * pkc[t * D + d];
    score[mm * KB + t] = s;
  }
  __syncthreads();

  // thread t owns row t: destructive top-6 (strict > keeps lowest index on tie,
  // matching jax.lax.top_k), then ascending index sort.
  float* row = score + t * KB;
  int idx[TOPK];
  for (int j = 0; j < TOPK; ++j) {
    float best = -3.4e38f;
    int bi = 0;
    for (int n = 0; n < KB; ++n) {
      const float val = row[n];
      if (val > best) { best = val; bi = n; }
    }
    idx[j] = bi;
    row[bi] = -3.4e38f;
  }
  for (int a = 0; a < TOPK - 1; ++a)
    for (int c = a + 1; c < TOPK; ++c)
      if (idx[c] < idx[a]) { int tmp = idx[a]; idx[a] = idx[c]; idx[c] = tmp; }
  int* lr = lut + ((size_t)bh * M + t) * TOPK;
  for (int j = 0; j < TOPK; ++j) lr[j] = idx[j];
}

// -------------------------------------- block-sparse flash attention (o_s)
__global__ __launch_bounds__(128) void sparse_attn_kernel(
    const float* __restrict__ q, const float* __restrict__ k,
    const float* __restrict__ v, const int* __restrict__ lut,
    float* __restrict__ out) {
  constexpr int LDQ = D + 4;      // f16 lds pitch
  constexpr int LDP = BLKK + 4;
  __shared__ _Float16 qsh[BLKQ * LDQ];
  __shared__ _Float16 ksh[BLKK * LDQ];
  __shared__ _Float16 vsh[BLKK * LDQ];
  __shared__ _Float16 psh[4 * 16 * LDP];

  const int gid = blockIdx.x;
  const int m = gid % M, bh = gid / M;
  const int b = bh / H, h = bh % H;
  const int tid = threadIdx.x;
  const int wave = tid >> 5, lane = tid & 31;
  const int lm = lane & 15, hh = lane >> 4;

  for (int i = tid; i < BLKQ * D; i += 128) {
    const int r = i >> 7, d = i & (D - 1);
    qsh[r * LDQ + d] =
        (_Float16)(q[(((size_t)b * L + (size_t)m * BLKQ + r) * H + h) * D + d] * SCALE);
  }
  __syncthreads();

  v16h aq[4];
#pragma unroll
  for (int kc = 0; kc < 4; ++kc) aq[kc] = load_A(qsh, LDQ, wave * 16, kc * 32);

  v8f o[8];
  float mrun[8], lrun[8];
#pragma unroll
  for (int j = 0; j < 8; ++j) o[j] = vzero();
#pragma unroll
  for (int r = 0; r < 8; ++r) { mrun[r] = -1e30f; lrun[r] = 0.f; }

  const int* lr = lut + ((size_t)bh * M + m) * TOPK;
  _Float16* pw = psh + wave * 16 * LDP;

  for (int t = 0; t < TOPK; ++t) {
    const int blk = lr[t];
    __syncthreads();  // previous tiles fully consumed before restaging
    const size_t kbase = (((size_t)b * L + (size_t)blk * BLKK) * H + h) * (size_t)D;
    for (int i = tid; i < BLKK * D; i += 128) {
      const int r = i >> 7, d = i & (D - 1);
      const size_t gi = kbase + (size_t)r * H * D + d;
      ksh[r * LDQ + d] = (_Float16)k[gi];
      vsh[r * LDQ + d] = (_Float16)v[gi];
    }
    if (t + 1 < TOPK) {  // prefetch next selected block's rows into caches
      const int nblk = lr[t + 1];
      const size_t nbase = (((size_t)b * L + (size_t)nblk * BLKK) * H + h) * (size_t)D;
      for (int i = tid; i < BLKK; i += 128) {
        __builtin_prefetch(k + nbase + (size_t)i * H * D, 0, 0);
        __builtin_prefetch(v + nbase + (size_t)i * H * D, 0, 0);
      }
    }
    __syncthreads();

    // S = (Q*scale) K^T : 16x64 per wave
    v8f s[4];
#pragma unroll
    for (int j = 0; j < 4; ++j) {
      v8f acc = vzero();
#pragma unroll
      for (int kc = 0; kc < 4; ++kc)
        acc = wmma16(aq[kc], load_Bt(ksh, LDQ, kc * 32, j * 16), acc);
      s[j] = acc;
    }

    // online softmax over the 64 new columns (row lives in 16-lane half)
    float mnew[8], corr[8], tsum[8];
#pragma unroll
    for (int r = 0; r < 8; ++r) {
      float tm = s[0][r];
#pragma unroll
      for (int j = 1; j < 4; ++j) tm = fmaxf(tm, s[j][r]);
#pragma unroll
      for (int msk = 1; msk < 16; msk <<= 1)
        tm = fmaxf(tm, __shfl_xor(tm, msk, 32));
      mnew[r] = fmaxf(mrun[r], tm);
      corr[r] = __expf(mrun[r] - mnew[r]);
      tsum[r] = 0.f;
    }
#pragma unroll
    for (int j = 0; j < 4; ++j) {
#pragma unroll
      for (int r = 0; r < 8; ++r) {
        const float p = __expf(s[j][r] - mnew[r]);
        tsum[r] += p;
        pw[(r + 8 * hh) * LDP + j * 16 + lm] = (_Float16)p;  // C->A relayout via LDS
      }
    }
#pragma unroll
    for (int r = 0; r < 8; ++r) {
      float ts = tsum[r];
#pragma unroll
      for (int msk = 1; msk < 16; msk <<= 1) ts += __shfl_xor(ts, msk, 32);
      lrun[r] = lrun[r] * corr[r] + ts;
      mrun[r] = mnew[r];
    }
#pragma unroll
    for (int j = 0; j < 8; ++j)
#pragma unroll
      for (int r = 0; r < 8; ++r) o[j][r] = o[j][r] * corr[r];

    // O += P V  (contraction 64 = 2 x K32)
    const v16h ap0 = load_A(pw, LDP, 0, 0);
    const v16h ap1 = load_A(pw, LDP, 0, 32);
#pragma unroll
    for (int j = 0; j < 8; ++j) {
      o[j] = wmma16(ap0, load_B(vsh, LDQ, 0, j * 16), o[j]);
      o[j] = wmma16(ap1, load_B(vsh, LDQ, 32, j * 16), o[j]);
    }
  }

#pragma unroll
  for (int r = 0; r < 8; ++r) {
    const float inv = 1.f / lrun[r];
    const int row = wave * 16 + r + 8 * hh;
    const size_t base =
        (((size_t)b * L + (size_t)m * BLKQ + row) * H + h) * (size_t)D;
#pragma unroll
    for (int j = 0; j < 8; ++j) out[base + j * 16 + lm] = o[j][r] * inv;
  }
}

// ------------------------------------ kvsum = k_fm^T @ v, ksum = colsum(k_fm)
__global__ __launch_bounds__(256) void kv_kernel(
    const float* __restrict__ k, const float* __restrict__ v,
    float* __restrict__ kvsum, float* __restrict__ ksum) {
  constexpr int LDC = D + 4;
  __shared__ _Float16 kf[32 * LDC];
  __shared__ _Float16 vf[32 * LDC];
  const int bh = blockIdx.x;
  const int b = bh / H, h = bh % H;
  const int tid = threadIdx.x;
  const int wave = tid >> 5, lane = tid & 31;
  const int lm = lane & 15, hh = lane >> 4;
  const int rl = tid >> 3, sub = tid & 7, c0 = sub * 16;

  v8f acc[8];
#pragma unroll
  for (int j = 0; j < 8; ++j) acc[j] = vzero();
  float kacc = 0.f;

  for (int c = 0; c < L / 32; ++c) {
    const int l = c * 32 + rl;
    const size_t gbase = (((size_t)b * L + l) * H + h) * (size_t)D;
    float x[16];
#pragma unroll
    for (int i = 0; i < 16; ++i) x[i] = k[gbase + c0 + i];
    float rmax = x[0];
#pragma unroll
    for (int i = 1; i < 16; ++i) rmax = fmaxf(rmax, x[i]);
#pragma unroll
    for (int msk = 1; msk < 8; msk <<= 1)
      rmax = fmaxf(rmax, __shfl_xor(rmax, msk, 32));
    float es = 0.f;
#pragma unroll
    for (int i = 0; i < 16; ++i) { x[i] = __expf(x[i] - rmax); es += x[i]; }
#pragma unroll
    for (int msk = 1; msk < 8; msk <<= 1) es += __shfl_xor(es, msk, 32);
    const float inv = 1.f / es;
#pragma unroll
    for (int i = 0; i < 16; ++i) {
      kf[rl * LDC + c0 + i] = (_Float16)(x[i] * inv);
      vf[rl * LDC + c0 + i] = (_Float16)v[gbase + c0 + i];
    }
    __syncthreads();

    if (tid < D) {
      float s = 0.f;
#pragma unroll
      for (int r = 0; r < 32; ++r) s += (float)kf[r * LDC + tid];
      kacc += s;
    }

    const v16h a = load_At(kf, LDC, wave * 16, 0);  // A[d][l] = kf[l][d]
#pragma unroll
    for (int j = 0; j < 8; ++j)
      acc[j] = wmma16(a, load_B(vf, LDC, 0, j * 16), acc[j]);
    __syncthreads();
  }

#pragma unroll
  for (int j = 0; j < 8; ++j)
#pragma unroll
    for (int r = 0; r < 8; ++r) {
      const int mrow = wave * 16 + r + 8 * hh;
      kvsum[((size_t)bh * D + mrow) * D + j * 16 + lm] = acc[j][r];
    }
  if (tid < D) ksum[(size_t)bh * D + tid] = kacc;
}

// -------------------------------------------------- kvW = kvsum @ W_proj^T
__global__ __launch_bounds__(256) void kvw_kernel(
    const float* __restrict__ kvsum, const float* __restrict__ W,
    float* __restrict__ kvW) {
  const int bh = blockIdx.x;
  for (int i = threadIdx.x; i < D * D; i += 256) {
    const int mm = i >> 7, n = i & (D - 1);
    const float* kr = kvsum + ((size_t)bh * D + mm) * D;
    const float* wr = W + (size_t)n * D;
    float s = 0.f;
    for (int e = 0; e < D; ++e) s += kr[e] * wr[e];
    kvW[(size_t)bh * D * D + i] = s;
  }
}

// -------------------- o += (q_fm @ kvW)/den + b  (den = eps + q_fm . ksum)
__global__ __launch_bounds__(256) void linear_out_kernel(
    const float* __restrict__ q, const float* __restrict__ kvW,
    const float* __restrict__ ksum, const float* __restrict__ bias,
    float* __restrict__ out) {
  constexpr int LDC = D + 4;
  constexpr int NLB = L / 64;
  __shared__ _Float16 qf[64 * LDC];
  __shared__ _Float16 kw[D * LDC];
  __shared__ float den[64];
  const int gid = blockIdx.x;
  const int lb = gid % NLB, bh = gid / NLB;
  const int b = bh / H, h = bh % H;
  const int tid = threadIdx.x;
  const int wave = tid >> 5, lane = tid & 31;
  const int lm = lane & 15, hh = lane >> 4;

  for (int i = tid; i < D * D; i += 256)
    kw[(i >> 7) * LDC + (i & (D - 1))] = (_Float16)kvW[(size_t)bh * D * D + i];

  {
    const int rl = tid >> 2, sub = tid & 3, cc = sub * 32;
    const int l = lb * 64 + rl;
    const size_t gbase = (((size_t)b * L + l) * H + h) * (size_t)D;
    float x[32];
#pragma unroll
    for (int i = 0; i < 32; ++i) x[i] = q[gbase + cc + i];
    float rmax = x[0];
#pragma unroll
    for (int i = 1; i < 32; ++i) rmax = fmaxf(rmax, x[i]);
#pragma unroll
    for (int msk = 1; msk < 4; msk <<= 1)
      rmax = fmaxf(rmax, __shfl_xor(rmax, msk, 32));
    float es = 0.f;
#pragma unroll
    for (int i = 0; i < 32; ++i) { x[i] = __expf(x[i] - rmax); es += x[i]; }
#pragma unroll
    for (int msk = 1; msk < 4; msk <<= 1) es += __shfl_xor(es, msk, 32);
    const float inv = 1.f / es;
    float dpart = 0.f;
    const float* ks = ksum + (size_t)bh * D + cc;
#pragma unroll
    for (int i = 0; i < 32; ++i) {
      const float fm = x[i] * inv;
      qf[rl * LDC + cc + i] = (_Float16)fm;
      dpart += fm * ks[i];
    }
#pragma unroll
    for (int msk = 1; msk < 4; msk <<= 1) dpart += __shfl_xor(dpart, msk, 32);
    if (sub == 0) den[rl] = EPSF + dpart;
  }
  __syncthreads();

  const int mtile = wave >> 1, nh = wave & 1;
  v16h aq[4];
#pragma unroll
  for (int kc = 0; kc < 4; ++kc) aq[kc] = load_A(qf, LDC, mtile * 16, kc * 32);

#pragma unroll
  for (int j = 0; j < 4; ++j) {
    v8f acc = vzero();
#pragma unroll
    for (int kc = 0; kc < 4; ++kc)
      acc = wmma16(aq[kc], load_B(kw, LDC, kc * 32, nh * 64 + j * 16), acc);
#pragma unroll
    for (int r = 0; r < 8; ++r) {
      const int row = mtile * 16 + r + 8 * hh;
      const int dcol = nh * 64 + j * 16 + lm;
      const size_t oi =
          (((size_t)b * L + (size_t)lb * 64 + row) * H + h) * (size_t)D + dcol;
      out[oi] += acc[r] / den[row] + bias[dcol];
    }
  }
}

}  // namespace sla

extern "C" void kernel_launch(void* const* d_in, const int* in_sizes, int n_in,
                              void* d_out, int out_size, void* d_ws, size_t ws_size,
                              hipStream_t stream) {
  using namespace sla;
  (void)in_sizes; (void)n_in; (void)out_size; (void)ws_size;

  const float* q  = (const float*)d_in[0];
  const float* k  = (const float*)d_in[1];
  const float* v  = (const float*)d_in[2];
  const float* W  = (const float*)d_in[3];
  const float* bp = (const float*)d_in[4];
  float* out = (float*)d_out;

  // workspace carve-up (~6.3 MB total)
  float* pq    = (float*)d_ws;
  float* pk    = pq + (size_t)B * H * M * D;
  int*   lut   = (int*)(pk + (size_t)B * H * KB * D);
  float* kvsum = (float*)(lut + (size_t)B * H * M * TOPK);
  float* kvW   = kvsum + (size_t)B * H * D * D;
  float* ksum  = kvW + (size_t)B * H * D * D;

  pool_kernel<<<B * H * M, 128, 0, stream>>>(q, k, pq, pk);
  route_kernel<<<B * H, 64, 0, stream>>>(pq, pk, lut);
  sparse_attn_kernel<<<B * H * M, 128, 0, stream>>>(q, k, v, lut, out);
  kv_kernel<<<B * H, 256, 0, stream>>>(k, v, kvsum, ksum);
  kvw_kernel<<<B * H, 256, 0, stream>>>(kvsum, W, kvW);
  linear_out_kernel<<<B * H * (L / 64), 256, 0, stream>>>(q, kvW, ksum, bp, out);
}
